// EfficientViTBlock_10153302687949
// MI455X (gfx1250) — matrix operand order repaired and checked by reading, not compiled, not run on hardware
//
#include <hip/hip_runtime.h>
#include <hip/hip_bf16.h>

typedef __attribute__((ext_vector_type(16))) __bf16 v16bf;
typedef __attribute__((ext_vector_type(8)))  float  v8f;
typedef __attribute__((ext_vector_type(4)))  unsigned int v4u;
typedef __attribute__((ext_vector_type(4)))  int   v4i;
typedef __attribute__((ext_vector_type(8)))  int   v8i;

#define BN_SCALE_F 0.9999950000374996f   // 1/sqrt(1+1e-5)
#define EPS_F 1e-15f

#define HW_C 4096          // H*W, fixed by the reference (64x64)
#define H_C  64
#define W_C  64

#if __has_builtin(__builtin_amdgcn_tensor_load_to_lds)
#define HAVE_TDM 1
#else
#define HAVE_TDM 0
#endif

__device__ __forceinline__ float hswish_f(float v) {
    float c = v + 3.0f;
    c = c < 0.0f ? 0.0f : (c > 6.0f ? 6.0f : c);
    return v * c * (1.0f / 6.0f);
}

#if HAVE_TDM
// ---------------------------------------------------------------------------
// TDM 2D tile load: rows x rowlen_dw dwords, global row stride gstride_elts
// dwords, into LDS with 4-dword padding after each 128-dword row
// (matches the [32][132] padded LDS tiles below). D# per ISA 08 §8.3/8.4.
// ---------------------------------------------------------------------------
__device__ __forceinline__ void tdm_load_2d(unsigned long long gaddr,
                                            unsigned int lds_addr,
                                            int rows, int rowlen_dw,
                                            int gstride_elts)
{
    v4u g0;
    g0.x = 1u;                                   // count=1, user descriptor
    g0.y = lds_addr;                             // LDS byte address
    g0.z = (unsigned int)gaddr;                  // global_addr[31:0]
    g0.w = (unsigned int)((gaddr >> 32) & 0x1FFFFFFull) | (2u << 30); // [56:32]|type=2
    v8i g1;
    g1[0] = (2 << 16)                            // data_size = 4 bytes
          | (1 << 20)                            // pad_enable
          | (6 << 22)                            // pad_interval: 128 dwords
          | (3 << 25);                           // pad_amount: 4 dwords
    g1[1] = (int)(((unsigned)gstride_elts & 0xFFFFu) << 16);   // tensor_dim0 lo16
    g1[2] = (int)((((unsigned)gstride_elts >> 16) & 0xFFFFu)   // tensor_dim0 hi16
          | ((unsigned)rows << 16));                           // tensor_dim1 lo16
    g1[3] = (int)((unsigned)rowlen_dw << 16);    // tensor_dim1 hi16=0 | tile_dim0
    g1[4] = rows;                                // tile_dim1 | tile_dim2=0
    g1[5] = gstride_elts;                        // tensor_dim0_stride[31:0]
    g1[6] = 0;                                   // stride hi | dim1_stride lo
    g1[7] = 0;
    v4i g2 = {0, 0, 0, 0};
    v4i g3 = {0, 0, 0, 0};
#if __clang_major__ <= 22
    __builtin_amdgcn_tensor_load_to_lds(g0, g1, g2, g3, 0);
#else
    v8i g4 = {0, 0, 0, 0, 0, 0, 0, 0};
    __builtin_amdgcn_tensor_load_to_lds(g0, g1, g2, g3, g4, 0);
#endif
}
#endif // HAVE_TDM

// ---------------------------------------------------------------------------
// Tiled bf16 WMMA GEMM over pixels:  Y[b, m, p] = act(sum_k W[m,k] * X[b,k,p])
// Tile 128M x 128N per workgroup, K stepped by 32, double-buffered LDS
// (ping-pong): global loads for tile k+1 are issued before WMMA of tile k,
// one barrier per K-step. Fragment-ordered LDS so per-wave fragment loads
// are ds_load_b128 pairs. 8 waves = 2(M) x 4(N); wave = 4x2 wmma tiles.
// ---------------------------------------------------------------------------
template<bool HAS_BIAS, bool HAS_BN, bool ACT_HSWISH, bool HAS_RES>
__global__ __launch_bounds__(256)
void gemm_wmma_k(const float* __restrict__ W, const float* __restrict__ X,
                 float* __restrict__ Y,
                 const float* __restrict__ bias,
                 const float* __restrict__ gamma, const float* __restrict__ beta,
                 const float* __restrict__ res,
                 int Cout, int Cin)
{
    constexpr int HW = HW_C;
    __shared__ __align__(32) __bf16 aT[2][8][32][16];   // 16 KB ping-pong
    __shared__ __align__(32) __bf16 bT[2][8][32][16];   // 16 KB ping-pong

    const int tid   = threadIdx.x;
    const int lane  = tid & 31;
    const int wv    = tid >> 5;          // wave 0..7
    const int wm    = wv >> 2;           // 0..1 (M)
    const int wn    = wv & 3;            // 0..3 (N)

    const int ntile = blockIdx.x;
    const int batch = ntile >> 5;                  // HW/128 = 32 tiles/batch
    const int nBase = (ntile & 31) << 7;
    const int mBase = blockIdx.y << 7;

    const float* Xb = X + (long)batch * Cin * HW;

    // staging roles
    const int sblk  = tid >> 5;          // block 0..7
    const int mLoc  = lane & 15;
    const int khalf = lane >> 4;

    // precomputed staging base pointers (k offset added per step)
    const float* wRow0 = W + (long)(mBase + sblk * 16 + mLoc) * Cin;
    const float* xCol0 = Xb + (long)(16 * khalf) * HW + nBase + sblk * 16 + mLoc;

    // A-fragment K mapping for this lane (contiguous: k = 16*khalf + i, matches B)
    int kmap[16];
    #pragma unroll
    for (int i = 0; i < 16; ++i) {
        kmap[i] = 16 * khalf + i;
    }

    v8f acc[4][2];
    const v8f vzero = {0.f,0.f,0.f,0.f,0.f,0.f,0.f,0.f};
    #pragma unroll
    for (int mi = 0; mi < 4; ++mi)
        #pragma unroll
        for (int ni = 0; ni < 2; ++ni)
            acc[mi][ni] = vzero;

    const int nK = Cin >> 5;

    // ---- prologue: stage tile 0 ------------------------------------------
    {
        float aS[16], bS[16];
        #pragma unroll
        for (int i = 0; i < 16; ++i) aS[i] = wRow0[kmap[i]];
        #pragma unroll
        for (int i = 0; i < 16; ++i) bS[i] = xCol0[i * HW];
        #pragma unroll
        for (int i = 0; i < 16; ++i) aT[0][sblk][lane][i] = (__bf16)aS[i];
        #pragma unroll
        for (int i = 0; i < 16; ++i) bT[0][sblk][lane][i] = (__bf16)bS[i];
    }
    __syncthreads();

    for (int kt = 0; kt < nK; ++kt) {
        const int cur = kt & 1;

        // ---- issue global loads for next tile (latency hidden by WMMA) ---
        float aN[16], bN[16];
        if (kt + 1 < nK) {
            const float* wRow = wRow0 + (kt + 1) * 32;
            const float* xCol = xCol0 + (long)((kt + 1) * 32) * HW;
            #pragma unroll
            for (int i = 0; i < 16; ++i) aN[i] = wRow[kmap[i]];
            #pragma unroll
            for (int i = 0; i < 16; ++i) bN[i] = xCol[i * HW];
        }
        if (kt + 2 < nK) {   // prefetch tile after next into caches
            __builtin_prefetch(wRow0 + (kt + 2) * 32, 0, 3);
            __builtin_prefetch(xCol0 + (long)((kt + 2) * 32) * HW, 0, 3);
        }

        // ---- fragments + 8 WMMAs on current buffer -----------------------
        v16bf af[4], bfr[2];
        #pragma unroll
        for (int mi = 0; mi < 4; ++mi)
            af[mi] = *reinterpret_cast<const v16bf*>(&aT[cur][wm * 4 + mi][lane][0]);
        #pragma unroll
        for (int ni = 0; ni < 2; ++ni)
            bfr[ni] = *reinterpret_cast<const v16bf*>(&bT[cur][wn * 2 + ni][lane][0]);

        #pragma unroll
        for (int mi = 0; mi < 4; ++mi)
            #pragma unroll
            for (int ni = 0; ni < 2; ++ni)
                acc[mi][ni] = __builtin_amdgcn_wmma_f32_16x16x32_bf16(
                    false, af[mi], false, bfr[ni],
                    (short)0, acc[mi][ni], false, false);

        // ---- store next tile to the other buffer -------------------------
        if (kt + 1 < nK) {
            const int nxt = cur ^ 1;
            #pragma unroll
            for (int i = 0; i < 16; ++i) aT[nxt][sblk][lane][i] = (__bf16)aN[i];
            #pragma unroll
            for (int i = 0; i < 16; ++i) bT[nxt][sblk][lane][i] = (__bf16)bN[i];
        }
        __syncthreads();
    }

    // ---- epilogue: C/D layout -> NCHW, fused bias/BN/hswish/residual -----
    float* Yb = Y + (long)batch * Cout * HW;
    const float* Rb = HAS_RES ? (res + (long)batch * Cout * HW) : nullptr;
    #pragma unroll
    for (int mi = 0; mi < 4; ++mi) {
        const int m0 = mBase + (wm * 4 + mi) * 16 + 8 * (lane >> 4);
        #pragma unroll
        for (int ni = 0; ni < 2; ++ni) {
            const int n = nBase + (wn * 2 + ni) * 16 + (lane & 15);
            float*       yp = Yb + (long)m0 * HW + n;
            const float* rp = HAS_RES ? (Rb + (long)m0 * HW + n) : nullptr;
            #pragma unroll
            for (int r = 0; r < 8; ++r) {
                const int m = m0 + r;
                float v = acc[mi][ni][r];
                if (HAS_BN)     v = gamma[m] * BN_SCALE_F * v + beta[m];
                if (HAS_BIAS)   v += bias[m];
                if (ACT_HSWISH) v = hswish_f(v);
                if (HAS_RES)    v += rp[r * HW];
                yp[r * HW] = v;
            }
        }
    }
}

// ---------------------------------------------------------------------------
// Depthwise KSxKS conv (zero pad), optional bias + hswish.
// ---------------------------------------------------------------------------
template<int KS, bool BIAS_ACT>
__global__ __launch_bounds__(256)
void dwconv_k(const float* __restrict__ in, const float* __restrict__ w,
              const float* __restrict__ bias, float* __restrict__ out,
              int C, long total)
{
    long idx = (long)blockIdx.x * 256 + threadIdx.x;
    if (idx >= total) return;
    constexpr int HW = HW_C, H = H_C, Wd = W_C;
    const int p  = (int)(idx & (HW - 1));
    const long bc = idx >> 12;
    const int c  = (int)(bc % C);
    const int yy = p >> 6, xx = p & 63;
    const float* inp = in + bc * HW;
    const float* wc  = w + c * KS * KS;
    constexpr int R = KS / 2;
    float s = 0.0f;
    #pragma unroll
    for (int ky = 0; ky < KS; ++ky) {
        int iy = yy + ky - R;
        if (iy < 0 || iy >= H) continue;
        #pragma unroll
        for (int kx = 0; kx < KS; ++kx) {
            int ix = xx + kx - R;
            if (ix < 0 || ix >= Wd) continue;
            s += wc[ky * KS + kx] * inp[iy * Wd + ix];
        }
    }
    if (BIAS_ACT) { s += bias[c]; s = hswish_f(s); }
    out[idx] = s;
}

// ---------------------------------------------------------------------------
// Grouped 32x32 pointwise: out[b, g*32+e, p] = sum_d W_pw[g,e,d]*in[b,g*32+d,p]
// ---------------------------------------------------------------------------
__global__ __launch_bounds__(256)
void pw_k(const float* __restrict__ in, const float* __restrict__ wpw,
          float* __restrict__ out, long total)
{
    long idx = (long)blockIdx.x * 256 + threadIdx.x;
    if (idx >= total) return;
    constexpr int HW = HW_C;
    const int p   = (int)(idx & (HW - 1));
    const long bch = idx >> 12;
    const int ch  = (int)(bch % 768);
    const long b  = bch / 768;
    const int g   = ch >> 5, e2 = ch & 31;
    const float* ip = in + (b * 768 + (long)g * 32) * HW + p;
    const float* wp = wpw + ((long)g * 32 + e2) * 32;
    float s = 0.0f;
    #pragma unroll
    for (int d = 0; d < 32; ++d) s += wp[d] * ip[d * HW];
    out[idx] = s;
}

// ---------------------------------------------------------------------------
// Linear attention, one workgroup per (batch, head).
// Phase 1: k/v tiles (32ch x 128n) staged into padded LDS — via the Tensor
//          Data Mover (tensor_load_to_lds + s_wait_tensorcnt) when the
//          builtin exists, else coalesced manual loads — then
//          vk[d,e] = sum_n v[d,n]*relu(k[e,n]) (+ ones row) -> LDS (33x32)
// Phase 2: out[d,n] = (vk[0:32]@q)[d,n] / ((vk[32]@q)[n] + eps)
// ---------------------------------------------------------------------------
__global__ __launch_bounds__(256)
void attn_k(const float* __restrict__ qkv, const float* __restrict__ ag,
            float* __restrict__ att)
{
    constexpr int HW = HW_C;
    constexpr int CH = 128;            // n-chunk
    constexpr int LP = CH + 4;         // padded stride (bank-conflict free,
                                       // matches TDM pad: 4 dw / 128 dw)
    __shared__ float kT[32][LP];       // ~16.5 KB
    __shared__ float vT[32][LP];       // ~16.5 KB
    __shared__ float vkS[33][32];

    const int bh = blockIdx.x;
    const int b  = bh >> 4;
    const int h  = bh & 15;
    const float* src = (h < 8) ? qkv : ag;
    const int ch0 = (h < 8 ? h : h - 8) * 96;
    const float* base = src + ((long)b * 768 + ch0) * HW;

    const int tid = threadIdx.x;
    const int e   = tid & 31;
    const int w   = tid >> 5;          // 0..7, each owns d = w*4..w*4+3

    float acc0 = 0.f, acc1 = 0.f, acc2 = 0.f, acc3 = 0.f, accOnes = 0.f;

    for (int n0 = 0; n0 < HW; n0 += CH) {
#if HAVE_TDM
        if (w == 0) {   // one wave drives the TDM; EXEC is ignored by TDM
            tdm_load_2d((unsigned long long)(base + 32 * HW + n0),
                        (unsigned int)(unsigned long long)&kT[0][0],
                        32, CH, HW);
            tdm_load_2d((unsigned long long)(base + 64 * HW + n0),
                        (unsigned int)(unsigned long long)&vT[0][0],
                        32, CH, HW);
            __builtin_amdgcn_s_wait_tensorcnt(0);
        }
        __syncthreads();
#else
        // coalesced staging: 32ch x 128n for k and v
        #pragma unroll
        for (int i = 0; i < 16; ++i) {
            const int lin = i * 256 + tid;           // 0..4095
            const int ch  = lin >> 7, nn = lin & 127;
            kT[ch][nn] = base[(32 + ch) * HW + n0 + nn];
            vT[ch][nn] = base[(64 + ch) * HW + n0 + nn];
        }
        __syncthreads();
#endif
        for (int nn = 0; nn < CH; ++nn) {
            float kv = kT[e][nn];
            kv = kv > 0.f ? kv : 0.f;
            acc0 += vT[w * 4 + 0][nn] * kv;
            acc1 += vT[w * 4 + 1][nn] * kv;
            acc2 += vT[w * 4 + 2][nn] * kv;
            acc3 += vT[w * 4 + 3][nn] * kv;
            if (w == 0) accOnes += kv;
        }
        __syncthreads();
    }
    vkS[w * 4 + 0][e] = acc0;
    vkS[w * 4 + 1][e] = acc1;
    vkS[w * 4 + 2][e] = acc2;
    vkS[w * 4 + 3][e] = acc3;
    if (w == 0) vkS[32][e] = accOnes;
    __syncthreads();

    float* attBase = att + ((long)b * 512 + h * 32) * HW;
    for (int n = tid; n < HW; n += 256) {
        float q[32];
        #pragma unroll
        for (int ee = 0; ee < 32; ++ee) {
            float t = base[ee * HW + n];
            q[ee] = t > 0.f ? t : 0.f;
        }
        float den = EPS_F;
        #pragma unroll
        for (int ee = 0; ee < 32; ++ee) den += vkS[32][ee] * q[ee];
        for (int d = 0; d < 32; ++d) {
            float num = 0.f;
            #pragma unroll
            for (int ee = 0; ee < 32; ++ee) num += vkS[d][ee] * q[ee];
            attBase[d * HW + n] = num / den;
        }
    }
}

// ---------------------------------------------------------------------------
extern "C" void kernel_launch(void* const* d_in, const int* in_sizes, int n_in,
                              void* d_out, int out_size, void* d_ws, size_t ws_size,
                              hipStream_t stream)
{
    const float* x      = (const float*)d_in[0];
    const float* W_qkv  = (const float*)d_in[1];
    const float* W_dw5  = (const float*)d_in[2];
    const float* W_pw   = (const float*)d_in[3];
    const float* W_proj = (const float*)d_in[4];
    const float* g_proj = (const float*)d_in[5];
    const float* b_proj = (const float*)d_in[6];
    const float* W_e    = (const float*)d_in[7];
    const float* b_e    = (const float*)d_in[8];
    const float* W_dw3  = (const float*)d_in[9];
    const float* b_dw3  = (const float*)d_in[10];
    const float* W_p    = (const float*)d_in[11];
    const float* g_p    = (const float*)d_in[12];
    const float* b_p    = (const float*)d_in[13];
    float* out = (float*)d_out;
    float* ws  = (float*)d_ws;

    const int B = 8;
    const int NT = (B * HW_C) / 128;         // 256 N-tiles

    // workspace layout (floats); reuse dead regions for e / d
    float* qkvB = ws;                        // 8*768*4096  = 25165824 (96 MB)
    float* dw5B = ws + 25165824L;            // 96 MB
    float* agB  = ws + 50331648L;            // 96 MB
    float* attB = ws + 75497472L;            // 8*512*4096 = 16777216 (64 MB)
    float* yB   = ws + 92274688L;            // 8*256*4096 =  8388608 (32 MB)
    float* eB   = ws;                        // reuse qkv+dw5 region (needs 128 MB)
    float* dB   = ws + 50331648L;            // reuse ag+att region  (needs 128 MB)

    dim3 blk(256);

    // 1. qkv = W_qkv @ x            (768 x 256)
    gemm_wmma_k<false, false, false, false><<<dim3(NT, 768 / 128), blk, 0, stream>>>(
        W_qkv, x, qkvB, nullptr, nullptr, nullptr, nullptr, 768, 256);

    // 2. dw5 depthwise 5x5 on qkv
    long t5 = (long)B * 768 * HW_C;
    dwconv_k<5, false><<<(unsigned)((t5 + 255) / 256), blk, 0, stream>>>(
        qkvB, W_dw5, nullptr, dw5B, 768, t5);

    // 3. grouped 32x32 pointwise -> ag
    pw_k<<<(unsigned)((t5 + 255) / 256), blk, 0, stream>>>(dw5B, W_pw, agB, t5);

    // 4. linear attention -> att (B, 512, HW)
    attn_k<<<B * 16, blk, 0, stream>>>(qkvB, agB, attB);

    // 5. y = BN(W_proj @ att) + x   (256 x 512)
    gemm_wmma_k<false, true, false, true><<<dim3(NT, 256 / 128), blk, 0, stream>>>(
        W_proj, attB, yB, nullptr, g_proj, b_proj, x, 256, 512);

    // 6. e = hswish(W_e @ y + b_e)  (1024 x 256)
    gemm_wmma_k<true, false, true, false><<<dim3(NT, 1024 / 128), blk, 0, stream>>>(
        W_e, yB, eB, b_e, nullptr, nullptr, nullptr, 1024, 256);

    // 7. d = hswish(dw3(e) + b_dw3)
    long t3 = (long)B * 1024 * HW_C;
    dwconv_k<3, true><<<(unsigned)((t3 + 255) / 256), blk, 0, stream>>>(
        eB, W_dw3, b_dw3, dB, 1024, t3);

    // 8. out = BN(W_p @ d) + y      (256 x 1024)
    gemm_wmma_k<false, true, false, true><<<dim3(NT, 256 / 128), blk, 0, stream>>>(
        W_p, dB, out, nullptr, g_p, b_p, yB, 256, 1024);
}
